// VSAWindowAttention_33646773796925
// MI455X (gfx1250) — matrix operand order repaired and checked by reading, not compiled, not run on hardware
//
#include <hip/hip_runtime.h>
#include <hip/hip_bf16.h>
#include <math.h>

typedef __attribute__((ext_vector_type(16))) _Float16 v16h;
typedef __attribute__((ext_vector_type(8)))  _Float16 v8h;   // 16-byte vector
typedef __attribute__((ext_vector_type(8)))  float    v8f;

#define BATCH 32
#define DIM   256
#define HEADS 8
#define WSZ   7
#define HH    56
#define WW    56
#define OHD   32
#define NWIN  64      // 8x8 windows
#define HW    3136    // 56*56
#define WTOK  49      // 7*7
#define ATT_SCALE 0.17677669529663689f  // 32^-0.5

static __device__ __forceinline__ int lane_id() { return threadIdx.x & 31; }

// ---------------- elementwise f32 -> f16 ----------------
__global__ void f32_to_f16_kernel(const float* __restrict__ src,
                                  _Float16* __restrict__ dst, long long n) {
  long long i = (long long)blockIdx.x * blockDim.x + threadIdx.x;
  long long stride = (long long)gridDim.x * blockDim.x;
  for (; i < n; i += stride) dst[i] = (_Float16)src[i];
}

// ---------------- avg-pool 7x7 + leaky relu ----------------
__global__ void pool_kernel(const float* __restrict__ x, float* __restrict__ pooled) {
  int idx = blockIdx.x * blockDim.x + threadIdx.x;     // B*DIM*64
  if (idx >= BATCH * DIM * 64) return;
  int ij = idx & 63;
  int c  = (idx >> 6) & 255;
  int b  = idx >> 14;
  int wi = ij >> 3, wj = ij & 7;
  const float* base = x + ((size_t)(b * DIM + c)) * HW + (wi * WSZ) * WW + wj * WSZ;
  float s = 0.f;
#pragma unroll
  for (int dy = 0; dy < WSZ; ++dy)
#pragma unroll
    for (int dx = 0; dx < WSZ; ++dx) s += base[dy * WW + dx];
  float v = s * (1.0f / 49.0f);
  pooled[idx] = (v > 0.f) ? v : 0.01f * v;
}

// ---------------- offset / scale 1x1 convs ----------------
__global__ void offscale_kernel(const float* __restrict__ pooled,
                                const float* __restrict__ w_off,
                                const float* __restrict__ b_off,
                                const float* __restrict__ w_scale,
                                const float* __restrict__ b_scale,
                                float* __restrict__ offws,
                                float* __restrict__ sclws) {
  int idx = blockIdx.x * blockDim.x + threadIdx.x;     // B*16*64
  if (idx >= BATCH * 16 * 64) return;
  int ij = idx & 63;
  int o  = (idx >> 6) & 15;
  int b  = idx >> 10;
  const float* pb = pooled + ((size_t)b * DIM) * 64 + ij;
  float so = b_off[o], ss = b_scale[o];
  for (int c = 0; c < DIM; ++c) {
    float pv = pb[(size_t)c * 64];
    so += w_off[o * DIM + c] * pv;
    ss += w_scale[o * DIM + c] * pv;
  }
  offws[idx] = so * 0.125f;   // / (W // WS) == / (H // WS) == 8
  sclws[idx] = ss;
}

// ---------------- WMMA fragment helpers (ISA layouts) ----------------
// A (16x32 f16): lane -> M = lane%16; element e: K = 16*(e/8) + ((e/2)%4)*2 + (e&1) + 8*(lane/16)
// Per lane this touches two contiguous 8-half chunks -> 2x ds_load_b128.
__device__ __forceinline__ v16h frag_A_from_lds(const _Float16* As, int mbase, int ld) {
  int lane = lane_id();
  int lo = lane & 15, hi = lane >> 4;
  v16h a;
#pragma unroll
  for (int e = 0; e < 16; ++e) {
    int r = e >> 1, j = e & 1;
    int k = 16 * (r >> 2) + 2 * (r & 3) + j + 8 * hi;
    a[e] = As[(mbase + lo) * ld + k];
  }
  return a;
}

// B (32x16 f16) from an LDS tile stored transposed [n][k] (leading dim ld):
// lane -> N = lane%16; element e: K = 8*(lane/16) + e + 8*(e/8)  -> two contiguous 8-half chunks.
__device__ __forceinline__ v16h frag_B_T(const _Float16* BsT, int nbase, int ld) {
  int lane = lane_id();
  int lo = lane & 15, hi = lane >> 4;
  v16h b;
#pragma unroll
  for (int e = 0; e < 16; ++e) b[e] = BsT[(nbase + lo) * ld + 8 * hi + e + 8 * (e >> 3)];
  return b;
}

// ---------------- qkv GEMM: (768x256) x (256x3136) per batch ----------------
__global__ void qkv_gemm_kernel(const _Float16* __restrict__ wq,   // 768x256
                                const _Float16* __restrict__ xh,   // B x 256 x 3136
                                const float* __restrict__ bqkv,    // 768
                                _Float16* __restrict__ qw,         // (B*64*8) x 49 x 32
                                _Float16* __restrict__ kfeat,      // (B*8) x 32 x 3136
                                _Float16* __restrict__ vfeat) {
  __shared__ __align__(32) _Float16 As[64 * 32];    // [m][k]
  __shared__ __align__(32) _Float16 BsT[64 * 32];   // [n][k]  (transposed)
  int n0 = blockIdx.x * 64;
  int m0 = blockIdx.y * 64;
  int b  = blockIdx.z;
  int t = threadIdx.x, wave = t >> 5, lane = t & 31;
  int lo = lane & 15, hi = lane >> 4;

  v8f acc[4] = {};
  for (int k0 = 0; k0 < DIM; k0 += 32) {
    {   // A: 64 rows x 32 halves of weights, vectorized copy
      int row = t >> 1, seg = (t & 1) * 16;
      const v8h* src = (const v8h*)(wq + (size_t)(m0 + row) * DIM + k0 + seg);
      v8h* dst = (v8h*)(As + row * 32 + seg);
      dst[0] = src[0];
      dst[1] = src[1];
    }
    {   // B: 32 k-rows x 64 cols, vector load + transpose scatter into BsT
      int row = t >> 2, seg = (t & 3) * 16;
      const v8h* src = (const v8h*)(xh + ((size_t)b * DIM + k0 + row) * HW + n0 + seg);
      v8h x0 = src[0], x1 = src[1];
#pragma unroll
      for (int u = 0; u < 8; ++u) BsT[(seg + u) * 32 + row] = x0[u];
#pragma unroll
      for (int u = 0; u < 8; ++u) BsT[(seg + 8 + u) * 32 + row] = x1[u];
    }
    __syncthreads();
    v16h a = frag_A_from_lds(As, wave * 16, 32);
#pragma unroll
    for (int nt = 0; nt < 4; ++nt) {
      v16h bf = frag_B_T(BsT, nt * 16, 32);
      acc[nt] = __builtin_amdgcn_wmma_f32_16x16x32_f16(false, a, false, bf,
                                                       (short)0, acc[nt], false, false);
    }
    __syncthreads();
  }
  // epilogue: route q -> windowed, k/v -> channel-major feature maps
#pragma unroll
  for (int nt = 0; nt < 4; ++nt) {
#pragma unroll
    for (int r = 0; r < 8; ++r) {
      int m = m0 + wave * 16 + r + 8 * hi;
      int n = n0 + nt * 16 + lo;
      float val = acc[nt][r] + bqkv[m];
      int sec  = m >> 8;
      int head = (m & 255) >> 5;
      int d    = m & 31;
      if (sec == 0) {
        int h = n / WW, w = n % WW;
        int win = (h / WSZ) * 8 + (w / WSZ);
        int wi  = (h % WSZ) * WSZ + (w % WSZ);
        qw[(((size_t)b * NWIN + win) * HEADS + head) * (WTOK * OHD) + wi * OHD + d] =
            (_Float16)val;
      } else {
        _Float16* dst = (sec == 1) ? kfeat : vfeat;
        dst[(((size_t)b * HEADS + head) * OHD + d) * HW + n] = (_Float16)val;
      }
    }
  }
}

// ---------------- deformable bilinear sampling of k/v ----------------
__global__ void grid_sample_kernel(const _Float16* __restrict__ kfeat,
                                   const _Float16* __restrict__ vfeat,
                                   const float* __restrict__ offws,
                                   const float* __restrict__ sclws,
                                   _Float16* __restrict__ kw,
                                   _Float16* __restrict__ vw) {
  int idx = blockIdx.x * blockDim.x + threadIdx.x;    // (B*HEADS) * 3136
  if (idx >= BATCH * HEADS * HW) return;
  int p = idx % HW;
  int n = idx / HW;
  int b = n >> 3, head = n & 7;
  int h = p / WW, w = p % WW;
  int wi = h / WSZ, wj = w / WSZ;             // window row / col
  int br = h % WSZ, bc = w % WSZ;             // intra-window row / col
  int ij = wi * 8 + wj;

  float sx = sclws[((size_t)b * 16 + head * 2 + 0) * 64 + ij];
  float sy = sclws[((size_t)b * 16 + head * 2 + 1) * 64 + ij];
  float ox = offws[((size_t)b * 16 + head * 2 + 0) * 64 + ij];
  float oy = offws[((size_t)b * 16 + head * 2 + 1) * 64 + ij];

  const float inv = 2.0f / 55.0f;
  float bw = (bc - 3) * inv, bh = (br - 3) * inv;
  float winx = -1.0f + (7 * wj + 3) * inv;
  float winy = -1.0f + (7 * wi + 3) * inv;
  float gx = winx + bw * (1.0f + sx) + ox;
  float gy = winy + bh * (1.0f + sy) + oy;

  float px = (gx + 1.0f) * 0.5f * (WW - 1);
  float py = (gy + 1.0f) * 0.5f * (HH - 1);
  float fx = floorf(px), fy = floorf(py);
  float wx = px - fx, wy = py - fy;
  int x0 = (int)fx, y0 = (int)fy, x1 = x0 + 1, y1 = y0 + 1;
  bool vx0 = (x0 >= 0) & (x0 < WW), vx1 = (x1 >= 0) & (x1 < WW);
  bool vy0 = (y0 >= 0) & (y0 < HH), vy1 = (y1 >= 0) & (y1 < HH);
  int xc0 = min(max(x0, 0), WW - 1), xc1 = min(max(x1, 0), WW - 1);
  int yc0 = min(max(y0, 0), HH - 1), yc1 = min(max(y1, 0), HH - 1);
  float w00 = (vx0 && vy0) ? (1 - wx) * (1 - wy) : 0.f;
  float w10 = (vx1 && vy0) ? wx * (1 - wy)       : 0.f;
  float w01 = (vx0 && vy1) ? (1 - wx) * wy       : 0.f;
  float w11 = (vx1 && vy1) ? wx * wy             : 0.f;
  int i00 = yc0 * WW + xc0, i10 = yc0 * WW + xc1;
  int i01 = yc1 * WW + xc0, i11 = yc1 * WW + xc1;

  size_t out_base = (((size_t)b * NWIN + ij) * HEADS + head) * (WTOK * OHD)
                    + (br * WSZ + bc) * OHD;
  size_t feat_base = (size_t)n * OHD * HW;
#pragma unroll 4
  for (int d = 0; d < OHD; ++d) {
    const _Float16* kc = kfeat + feat_base + (size_t)d * HW;
    const _Float16* vc = vfeat + feat_base + (size_t)d * HW;
    float kv = w00 * (float)kc[i00] + w10 * (float)kc[i10]
             + w01 * (float)kc[i01] + w11 * (float)kc[i11];
    float vv = w00 * (float)vc[i00] + w10 * (float)vc[i10]
             + w01 * (float)vc[i01] + w11 * (float)vc[i11];
    kw[out_base + d] = (_Float16)kv;
    vw[out_base + d] = (_Float16)vv;
  }
}

// ---------------- per head-window attention (WMMA) ----------------
__global__ void attention_kernel(const _Float16* __restrict__ qw,
                                 const _Float16* __restrict__ kw,
                                 const _Float16* __restrict__ vw,
                                 const float* __restrict__ rpb,   // 169 x 8
                                 _Float16* __restrict__ ows) {
  __shared__ __align__(32) _Float16 qs [64 * 32];   // [tok][d]
  __shared__ __align__(32) _Float16 ks [64 * 32];   // [tok][d]
  __shared__ __align__(32) _Float16 vsT[32 * 64];   // [d][tok]  (transposed)
  __shared__ __align__(32) float    ds [64 * 64];
  __shared__ __align__(32) _Float16 as2[64 * 64];

  int wh = blockIdx.x;                    // (b*64+win)*8 + head
  int head = wh & 7;
  int t = threadIdx.x, wave = t >> 5, lane = t & 31;
  int lo = lane & 15, hi = lane >> 4;
  size_t base = (size_t)wh * (WTOK * OHD);

  // vectorized load: 1568 halves = 196 chunks of 8; pad chunks 196..255 with 0
  const v8h* q8 = (const v8h*)(qw + base);
  const v8h* k8 = (const v8h*)(kw + base);
  const v8h* v8 = (const v8h*)(vw + base);
  for (int c = t; c < 256; c += 128) {
    v8h zq = {}, zk = {}, zv = {};
    if (c < 196) { zq = q8[c]; zk = k8[c]; zv = v8[c]; }
    ((v8h*)qs)[c] = zq;
    ((v8h*)ks)[c] = zk;
    int tok = c >> 2;            // (c*8)/32
    int d0  = (c & 3) * 8;
#pragma unroll
    for (int u = 0; u < 8; ++u) vsT[(d0 + u) * 64 + tok] = zv[u];
  }
  __syncthreads();

  // dots = q @ k^T  (64x64, one 16-row strip per wave)
  int m0 = wave * 16;
  {
    v16h a = frag_A_from_lds(qs, m0, 32);
#pragma unroll
    for (int nt = 0; nt < 4; ++nt) {
      v16h bf = frag_B_T(ks, nt * 16, 32);   // k is [tok][d] == B^T already
      v8f c = {};
      c = __builtin_amdgcn_wmma_f32_16x16x32_f16(false, a, false, bf,
                                                 (short)0, c, false, false);
#pragma unroll
      for (int r = 0; r < 8; ++r) {
        int m = m0 + r + 8 * hi;
        int n = nt * 16 + lo;
        float v = c[r] * ATT_SCALE;
        if (n < WTOK) {
          if (m < WTOK) {
            int dy = (m / WSZ) - (n / WSZ);
            int dx = (m % WSZ) - (n % WSZ);
            v += rpb[((dy + 6) * 13 + (dx + 6)) * HEADS + head];
          }
        } else {
          v = -1e30f;
        }
        ds[m * 64 + n] = v;
      }
    }
  }
  __syncthreads();

  // softmax: one row per thread (rows 0..63)
  if (t < 64) {
    float mx = -1e30f;
    for (int c = 0; c < 64; ++c) mx = fmaxf(mx, ds[t * 64 + c]);
    float s = 0.f;
    for (int c = 0; c < 64; ++c) {
      float e = __expf(ds[t * 64 + c] - mx);
      ds[t * 64 + c] = e;
      s += e;
    }
    float invs = 1.0f / s;
    for (int c = 0; c < 64; ++c) as2[t * 64 + c] = (_Float16)(ds[t * 64 + c] * invs);
  }
  __syncthreads();

  // out = attn(64x64) @ v(64x32)
  v8f oacc[2] = {};
#pragma unroll
  for (int k0 = 0; k0 < 64; k0 += 32) {
    v16h a;
#pragma unroll
    for (int e = 0; e < 16; ++e) {
      int r = e >> 1, j = e & 1;
      int k = 16 * (r >> 2) + 2 * (r & 3) + j + 8 * hi;
      a[e] = as2[(m0 + lo) * 64 + k0 + k];
    }
#pragma unroll
    for (int nt = 0; nt < 2; ++nt) {
      v16h bf;
#pragma unroll
      for (int e = 0; e < 16; ++e)
        bf[e] = vsT[(nt * 16 + lo) * 64 + k0 + 8 * hi + e + 8 * (e >> 3)];   // two 8-half chunks
      oacc[nt] = __builtin_amdgcn_wmma_f32_16x16x32_f16(false, a, false, bf,
                                                        (short)0, oacc[nt], false, false);
    }
  }
#pragma unroll
  for (int nt = 0; nt < 2; ++nt)
#pragma unroll
    for (int r = 0; r < 8; ++r) {
      int m = m0 + r + 8 * hi;
      if (m < WTOK)
        ows[base + m * OHD + nt * 16 + lo] = (_Float16)oacc[nt][r];
    }
}

// ---------------- windows -> (B, 256, 56, 56) f16 ----------------
__global__ void windows_to_feat_kernel(const _Float16* __restrict__ ows,
                                       _Float16* __restrict__ feat) {
  long long idx = (long long)blockIdx.x * blockDim.x + threadIdx.x;
  if (idx >= (long long)BATCH * DIM * HW) return;
  int p = (int)(idx % HW);
  int o = (int)((idx / HW) % DIM);
  int b = (int)(idx / ((long long)DIM * HW));
  int head = o >> 5, d = o & 31;
  int h = p / WW, w = p % WW;
  int win = (h / WSZ) * 8 + (w / WSZ);
  int wi  = (h % WSZ) * WSZ + (w % WSZ);
  feat[idx] = ows[(((size_t)b * NWIN + win) * HEADS + head) * (WTOK * OHD) + wi * OHD + d];
}

// ---------------- proj GEMM: (256x256) x (256x3136) per batch, f32 out ------
__global__ void proj_gemm_kernel(const _Float16* __restrict__ wp,   // 256x256
                                 const _Float16* __restrict__ feat, // B x 256 x 3136
                                 const float* __restrict__ bproj,
                                 float* __restrict__ out) {
  __shared__ __align__(32) _Float16 As[64 * 32];    // [m][k]
  __shared__ __align__(32) _Float16 BsT[64 * 32];   // [n][k]
  int n0 = blockIdx.x * 64;
  int m0 = blockIdx.y * 64;
  int b  = blockIdx.z;
  int t = threadIdx.x, wave = t >> 5, lane = t & 31;
  int lo = lane & 15, hi = lane >> 4;

  v8f acc[4] = {};
  for (int k0 = 0; k0 < DIM; k0 += 32) {
    {
      int row = t >> 1, seg = (t & 1) * 16;
      const v8h* src = (const v8h*)(wp + (size_t)(m0 + row) * DIM + k0 + seg);
      v8h* dst = (v8h*)(As + row * 32 + seg);
      dst[0] = src[0];
      dst[1] = src[1];
    }
    {
      int row = t >> 2, seg = (t & 3) * 16;
      const v8h* src = (const v8h*)(feat + ((size_t)b * DIM + k0 + row) * HW + n0 + seg);
      v8h x0 = src[0], x1 = src[1];
#pragma unroll
      for (int u = 0; u < 8; ++u) BsT[(seg + u) * 32 + row] = x0[u];
#pragma unroll
      for (int u = 0; u < 8; ++u) BsT[(seg + 8 + u) * 32 + row] = x1[u];
    }
    __syncthreads();
    v16h a = frag_A_from_lds(As, wave * 16, 32);
#pragma unroll
    for (int nt = 0; nt < 4; ++nt) {
      v16h bf = frag_B_T(BsT, nt * 16, 32);
      acc[nt] = __builtin_amdgcn_wmma_f32_16x16x32_f16(false, a, false, bf,
                                                       (short)0, acc[nt], false, false);
    }
    __syncthreads();
  }
#pragma unroll
  for (int nt = 0; nt < 4; ++nt)
#pragma unroll
    for (int r = 0; r < 8; ++r) {
      int m = m0 + wave * 16 + r + 8 * hi;
      int n = n0 + nt * 16 + lo;
      out[((size_t)b * DIM + m) * HW + n] = acc[nt][r] + bproj[m];
    }
}

// ---------------- launcher ----------------
extern "C" void kernel_launch(void* const* d_in, const int* in_sizes, int n_in,
                              void* d_out, int out_size, void* d_ws, size_t ws_size,
                              hipStream_t stream) {
  const float* x       = (const float*)d_in[0];
  const float* w_off   = (const float*)d_in[1];
  const float* b_off   = (const float*)d_in[2];
  const float* w_scale = (const float*)d_in[3];
  const float* b_scale = (const float*)d_in[4];
  const float* w_qkv   = (const float*)d_in[5];
  const float* b_qkv   = (const float*)d_in[6];
  const float* w_proj  = (const float*)d_in[7];
  const float* b_proj  = (const float*)d_in[8];
  const float* rpb     = (const float*)d_in[9];
  float* out = (float*)d_out;

  const size_t NELEM = (size_t)BATCH * DIM * HW;          // 25,690,112
  const size_t BIG   = ((NELEM * sizeof(_Float16) + 255) & ~(size_t)255);

  char* ws = (char*)d_ws;
  _Float16* xh     = (_Float16*)(ws + 0 * BIG);   // later reused as kw
  _Float16* kfeat  = (_Float16*)(ws + 1 * BIG);   // later reused as ows
  _Float16* vfeat  = (_Float16*)(ws + 2 * BIG);   // later reused as attn feat
  _Float16* qw     = (_Float16*)(ws + 3 * BIG);
  _Float16* vw     = (_Float16*)(ws + 4 * BIG);
  char* small = ws + 5 * BIG;
  _Float16* wqkvh  = (_Float16*)small;            small += (768 * 256 * 2 + 255) & ~255;
  _Float16* wprojh = (_Float16*)small;            small += (256 * 256 * 2 + 255) & ~255;
  float* pooled    = (float*)small;               small += (BATCH * DIM * 64 * 4 + 255) & ~255;
  float* offws     = (float*)small;               small += (BATCH * 16 * 64 * 4 + 255) & ~255;
  float* sclws     = (float*)small;               small += (BATCH * 16 * 64 * 4 + 255) & ~255;

  _Float16* kw       = xh;      // safe: xh consumed by qkv gemm before kw written
  _Float16* ows      = kfeat;   // safe: kfeat consumed by grid_sample before attn writes
  _Float16* attnfeat = vfeat;   // safe: vfeat consumed by grid_sample

  // 1) conversions
  f32_to_f16_kernel<<<4096, 256, 0, stream>>>(x, xh, (long long)NELEM);
  f32_to_f16_kernel<<<192, 256, 0, stream>>>(w_qkv, wqkvh, 768LL * 256);
  f32_to_f16_kernel<<<64, 256, 0, stream>>>(w_proj, wprojh, 256LL * 256);

  // 2) pooling + offsets/scales
  pool_kernel<<<(BATCH * DIM * 64 + 255) / 256, 256, 0, stream>>>(x, pooled);
  offscale_kernel<<<(BATCH * 16 * 64 + 255) / 256, 256, 0, stream>>>(
      pooled, w_off, b_off, w_scale, b_scale, offws, sclws);

  // 3) qkv GEMM (WMMA)
  qkv_gemm_kernel<<<dim3(HW / 64, 768 / 64, BATCH), 128, 0, stream>>>(
      wqkvh, xh, b_qkv, qw, kfeat, vfeat);

  // 4) deformable sampling of k/v
  grid_sample_kernel<<<(BATCH * HEADS * HW + 255) / 256, 256, 0, stream>>>(
      kfeat, vfeat, offws, sclws, kw, vw);

  // 5) attention (WMMA)
  attention_kernel<<<BATCH * NWIN * HEADS, 128, 0, stream>>>(qw, kw, vw, rpb, ows);

  // 6) rearrange + proj GEMM (WMMA)
  windows_to_feat_kernel<<<(int)((NELEM + 255) / 256), 256, 0, stream>>>(ows, attnfeat);
  proj_gemm_kernel<<<dim3(HW / 64, DIM / 64, BATCH), 128, 0, stream>>>(
      wprojh, attnfeat, b_proj, out);
}
